// MultiheadAttention_70720931496596
// MI455X (gfx1250) — compile-verified, matrix operand order checked by reading, not run on hardware
//
#include <hip/hip_runtime.h>
#include <hip/hip_bf16.h>

// ---------------------------------------------------------------------------
// MI455X (gfx1250) multi-head attention, bf16 WMMA pipeline:
//   0) convert Wq/Wk/Wv/Wo (and, if ws allows, q/k/v) f32 -> bf16
//   1) gemm mode 0: Qh = q @ Wq^T + bq   -> bf16 [B*H, S, 64]
//      gemm mode 0: Kh = k @ Wk^T + bk   -> bf16 [B*H, S, 64]
//      gemm mode 2: Vt = (v @ Wv^T + bv)^T per head -> bf16 [B*H, 64, S]
//      (B-tile staged to LDS with GLOBAL_LOAD_ASYNC_TO_LDS_B128, double-buffered)
//   2) flash attention (online softmax): K/V tiles async-staged to LDS,
//      WMMA QK^T and PV, P transposed via LDS round-trip
//      -> Abuf bf16 [B*S, 1024] (token-major, feature = h*64+d)
//   3) gemm mode 1: out = Abuf @ Wo^T + bo -> f32 d_out
// ---------------------------------------------------------------------------

typedef __attribute__((ext_vector_type(16))) __bf16 v16bf;
typedef __attribute__((ext_vector_type(4)))  __bf16 v4bf;
typedef __attribute__((ext_vector_type(8)))  float  v8f;

#define EMB_   1024
#define SEQ_   2048
#define NHEAD_ 16
#define DK_    64
#define BATCH_ 4
#define MROWS_ (BATCH_ * SEQ_)   // 8192

__device__ __forceinline__ v8f wmma_bf16(v16bf a, v16bf b, v8f c) {
  // D = A(16x32) * B(32x16) + C, f32 accumulate
  return __builtin_amdgcn_wmma_f32_16x16x32_bf16(
      /*neg_a=*/false, a, /*neg_b=*/false, b,
      /*c_mod=*/(short)0, c, /*reuse_a=*/false, /*reuse_b=*/false);
}

// Load a 16-element bf16 fragment slice: 8 contiguous at p, 8 at p+16.
// Works for global (global_load_b128) and LDS (ds_load_b128) pointers.
__device__ __forceinline__ v16bf ldbf16x16(const __bf16* p) {
  union { v16bf v; uint4 q[2]; } u;
  u.q[0] = *(const uint4*)(p);
  u.q[1] = *(const uint4*)(p + 16);
  return u.v;
}

// Convert 16 f32 (two groups of 8: p[0..7], p[16..23]) to a bf16 fragment.
__device__ __forceinline__ v16bf cvt16(const float* p) {
  float4 a = *(const float4*)(p);
  float4 b = *(const float4*)(p + 4);
  float4 c = *(const float4*)(p + 16);
  float4 d = *(const float4*)(p + 20);
  v16bf r;
  r[0]  = (__bf16)a.x; r[1]  = (__bf16)a.y; r[2]  = (__bf16)a.z; r[3]  = (__bf16)a.w;
  r[4]  = (__bf16)b.x; r[5]  = (__bf16)b.y; r[6]  = (__bf16)b.z; r[7]  = (__bf16)b.w;
  r[8]  = (__bf16)c.x; r[9]  = (__bf16)c.y; r[10] = (__bf16)c.z; r[11] = (__bf16)c.w;
  r[12] = (__bf16)d.x; r[13] = (__bf16)d.y; r[14] = (__bf16)d.z; r[15] = (__bf16)d.w;
  return r;
}

// ---------------------------------------------------------------------------
// f32 -> bf16 pre-conversion, 4 elements per thread.
// ---------------------------------------------------------------------------
__global__ __launch_bounds__(256)
void cvt_f32_to_bf16(const float* __restrict__ in, __bf16* __restrict__ out, int n4) {
  const int i = blockIdx.x * blockDim.x + threadIdx.x;
  if (i < n4) {
    const float4 f = ((const float4*)in)[i];
    v4bf o;
    o[0] = (__bf16)f.x; o[1] = (__bf16)f.y; o[2] = (__bf16)f.z; o[3] = (__bf16)f.w;
    ((v4bf*)out)[i] = o;
  }
}

// ---------------------------------------------------------------------------
// GEMM: Y[m,n] = sum_k A[m,k] * W[n,k] + bias[n],  M=8192, N=K=1024
// W is bf16 [N,K]; its 128x32 block tile is staged to LDS via
// GLOBAL_LOAD_ASYNC_TO_LDS_B128 (double-buffered, ASYNCcnt-synchronized).
// mode 0: write bf16 head-split     out[((b*H+h)*S + s)*64 + d]
// mode 2: write bf16 head-split V^T out[((b*H+h)*64 + d)*S + s]
// mode 1: write f32 plain           out[m*1024 + n]
// a_bf16: A is bf16 [M,K] (else f32, converted on the fly)
// Block: 256 threads = 8 waves; wave tile 32(M) x 64(N); block tile 128x128.
// ---------------------------------------------------------------------------
__global__ __launch_bounds__(256)
void mha_gemm_wmma(const void* __restrict__ Ain, const __bf16* __restrict__ Wbf,
                   const float* __restrict__ bias, void* __restrict__ out,
                   int mode, int a_bf16) {
  constexpr int K = EMB_;
  __shared__ __bf16 tileB[2][128 * 32];          // 2 x 8 KB double buffer

  const int tid  = threadIdx.x;
  const int lane = tid & 31;
  const int wave = tid >> 5;
  const int l15  = lane & 15;
  const int hi   = lane >> 4;          // 0: lanes 0-15, 1: lanes 16-31
  const int mBase  = blockIdx.y * 128 + (wave & 3) * 32;
  const int nBlock = blockIdx.x * 128;
  const int wn     = wave >> 2;        // 0..1 -> 64-wide N slice within block

  const float*  Af  = (const float*)Ain;
  const __bf16* Abf = (const __bf16*)Ain;

  // LDS byte offsets of the two buffers (generic->LDS aperture: low 32 bits).
  const uint32_t ldsB0 = (uint32_t)(uintptr_t)(&tileB[0][0]);
  const uint32_t ldsB1 = (uint32_t)(uintptr_t)(&tileB[1][0]);

  // Stage one 128(cols) x 32(k) bf16 tile of W into LDS buffer `ldsBase`.
  // 512 16-byte chunks; each of 256 threads issues 2 async loads.
  auto stageB = [&](int k0, uint32_t ldsBase) {
#pragma unroll
    for (int p = 0; p < 2; ++p) {
      const int chunk = tid * 2 + p;           // 0..511
      const int col   = chunk >> 2;            // 0..127
      const int ko    = (chunk & 3) * 8;       // 0,8,16,24
      const uint32_t lds  = ldsBase + (uint32_t)(col * 32 + ko) * 2u;
      const uint32_t goff = (uint32_t)(((nBlock + col) * K + k0 + ko) * 2);
      asm volatile("global_load_async_to_lds_b128 %0, %1, %2"
                   :: "v"(lds), "v"(goff), "s"(Wbf)
                   : "memory");
    }
  };

  v8f acc[2][4] = {};

  stageB(0, ldsB0);
  int buf = 0;

  for (int k0 = 0; k0 < K; k0 += 32) {
    const bool more = (k0 + 32 < K);
    if (more) {
      stageB(k0 + 32, buf ? ldsB0 : ldsB1);
      asm volatile("s_wait_asynccnt 2" ::: "memory");   // current buf done
    } else {
      asm volatile("s_wait_asynccnt 0" ::: "memory");
    }
    __syncthreads();                                    // all waves staged

    // A fragments (per-wave, direct from global; prefetch next k-slice)
    v16bf af[2];
#pragma unroll
    for (int i = 0; i < 2; ++i) {
      const size_t row = (size_t)(mBase + i * 16 + l15);
      if (a_bf16) {
        const __bf16* pa = Abf + row * K + k0 + hi * 8;
        af[i] = ldbf16x16(pa);
        __builtin_prefetch(pa + 32, 0, 0);
      } else {
        const float* pa = Af + row * K + k0 + hi * 8;
        af[i] = cvt16(pa);
        __builtin_prefetch(pa + 32, 0, 0);
      }
    }

    // B fragments from LDS (ds_load_b128 x2 each)
    const __bf16* Bt = &tileB[buf][0];
    v16bf bf[4];
#pragma unroll
    for (int j = 0; j < 4; ++j)
      bf[j] = ldbf16x16(Bt + (wn * 64 + j * 16 + l15) * 32 + hi * 8);

#pragma unroll
    for (int i = 0; i < 2; ++i)
#pragma unroll
      for (int j = 0; j < 4; ++j)
        acc[i][j] = wmma_bf16(af[i], bf[j], acc[i][j]);

    __syncthreads();                 // reads done before next stage overwrites
    buf ^= 1;
  }

  // Epilogue. C/D layout: VGPR r holds row (r + 8*hi), col = lane&15.
#pragma unroll
  for (int i = 0; i < 2; ++i) {
#pragma unroll
    for (int j = 0; j < 4; ++j) {
      const int col = nBlock + wn * 64 + j * 16 + l15;
      const float bv = bias[col];
#pragma unroll
      for (int r = 0; r < 8; ++r) {
        const int row = mBase + i * 16 + r + 8 * hi;
        const float val = acc[i][j][r] + bv;
        if (mode == 1) {
          ((float*)out)[(size_t)row * EMB_ + col] = val;
        } else {
          const int b = row / SEQ_, s = row % SEQ_;
          const int h = col >> 6,  d = col & 63;
          size_t idx;
          if (mode == 0) idx = ((size_t)((b * NHEAD_ + h) * SEQ_) + s) * DK_ + d;
          else           idx = ((size_t)((b * NHEAD_ + h) * DK_) + d) * SEQ_ + s;
          ((__bf16*)out)[idx] = (__bf16)val;
        }
      }
    }
  }
}

// ---------------------------------------------------------------------------
// Flash attention: per (b,h), O = softmax(Q K^T / 8) V, online softmax.
// Block: 128 threads = 4 waves; wave owns 16 query rows x d=64.
// K tile [32 keys][64 d] and V^T tile [64 d][32 keys] are shared by all 4
// waves -> async-staged to LDS, double-buffered (4 async b128 per thread per
// stage, s_wait_asynccnt-synchronized).
// ---------------------------------------------------------------------------
__global__ __launch_bounds__(128)
void mha_flash_wmma(const __bf16* __restrict__ Qh, const __bf16* __restrict__ Kh,
                    const __bf16* __restrict__ Vt, __bf16* __restrict__ Abuf) {
  __shared__ __bf16 tileK[2][32 * 64];  // [key][d]   4 KB each
  __shared__ __bf16 tileV[2][64 * 32];  // [d][key]   4 KB each
  __shared__ __bf16 ldsP[4][16 * 32];   // per-wave P staging (1 KB each)

  const int tid  = threadIdx.x;
  const int lane = tid & 31;
  const int wave = tid >> 5;
  const int l15  = lane & 15;
  const int hi   = lane >> 4;
  const int bh   = blockIdx.y;                        // 0..63 = b*16+h
  const int q0   = blockIdx.x * 64 + wave * 16;

  const __bf16* Qb = Qh + (size_t)bh * SEQ_ * DK_;
  const __bf16* Kb = Kh + (size_t)bh * SEQ_ * DK_;
  const __bf16* Vb = Vt + (size_t)bh * DK_ * SEQ_;

  const uint32_t ldsK0 = (uint32_t)(uintptr_t)(&tileK[0][0]);
  const uint32_t ldsK1 = (uint32_t)(uintptr_t)(&tileK[1][0]);
  const uint32_t ldsV0 = (uint32_t)(uintptr_t)(&tileV[0][0]);
  const uint32_t ldsV1 = (uint32_t)(uintptr_t)(&tileV[1][0]);

  // Stage K (32x64) and V^T (64x32) tiles for key block k0 into buffer `b`.
  // K: 256 chunks of 16B; V: 256 chunks of 16B; 128 threads x (2+2) asyncs.
  auto stageKV = [&](int k0, uint32_t ldsK, uint32_t ldsV) {
#pragma unroll
    for (int p = 0; p < 2; ++p) {
      const int ck   = tid * 2 + p;           // 0..255
      const int key  = ck >> 3;               // 0..31
      const int dc   = (ck & 7) * 8;          // 0..56
      const uint32_t ldsa = ldsK + (uint32_t)(key * 64 + dc) * 2u;
      const uint32_t goff = (uint32_t)(((k0 + key) * DK_ + dc) * 2);
      asm volatile("global_load_async_to_lds_b128 %0, %1, %2"
                   :: "v"(ldsa), "v"(goff), "s"(Kb)
                   : "memory");
    }
#pragma unroll
    for (int p = 0; p < 2; ++p) {
      const int cv   = tid * 2 + p;           // 0..255
      const int drow = cv >> 2;               // 0..63
      const int kc   = (cv & 3) * 8;          // 0..24
      const uint32_t ldsa = ldsV + (uint32_t)(drow * 32 + kc) * 2u;
      const uint32_t goff = (uint32_t)((drow * SEQ_ + k0 + kc) * 2);
      asm volatile("global_load_async_to_lds_b128 %0, %1, %2"
                   :: "v"(ldsa), "v"(goff), "s"(Vb)
                   : "memory");
    }
  };

  // Q fragments (reused across the whole key loop): d 0..31 and 32..63
  v16bf qf[2];
#pragma unroll
  for (int db = 0; db < 2; ++db)
    qf[db] = ldbf16x16(Qb + (size_t)(q0 + l15) * DK_ + db * 32 + hi * 8);

  float m[8], l[8];
#pragma unroll
  for (int r = 0; r < 8; ++r) { m[r] = -1e30f; l[r] = 0.0f; }
  v8f acc[4] = {};

  stageKV(0, ldsK0, ldsV0);
  int buf = 0;

  for (int k0 = 0; k0 < SEQ_; k0 += 32) {
    const bool more = (k0 + 32 < SEQ_);
    if (more) {
      stageKV(k0 + 32, buf ? ldsK0 : ldsK1, buf ? ldsV0 : ldsV1);
      asm volatile("s_wait_asynccnt 4" ::: "memory");   // current tiles done
    } else {
      asm volatile("s_wait_asynccnt 0" ::: "memory");
    }
    __syncthreads();

    const __bf16* Kt = &tileK[buf][0];
    const __bf16* Vt2 = &tileV[buf][0];

    // ---- scores: two 16x16 tiles, accumulate over d=64 (2 WMMAs each) ----
    v8f s[2] = {};
#pragma unroll
    for (int j = 0; j < 2; ++j) {
#pragma unroll
      for (int db = 0; db < 2; ++db) {
        v16bf kf = ldbf16x16(Kt + (j * 16 + l15) * 64 + db * 32 + hi * 8);
        s[j] = wmma_bf16(qf[db], kf, s[j]);
      }
    }

    // ---- online softmax per row (VGPR slot r -> row r + 8*hi) ----
#pragma unroll
    for (int r = 0; r < 8; ++r) {
      float v0 = s[0][r] * 0.125f;        // 1/sqrt(64)
      float v1 = s[1][r] * 0.125f;
      float mx = fmaxf(v0, v1);
#pragma unroll
      for (int off = 1; off <= 8; off <<= 1)
        mx = fmaxf(mx, __shfl_xor(mx, off, 32));   // reduce within 16-lane half
      const float mnew = fmaxf(m[r], mx);
      const float corr = __expf(m[r] - mnew);
      const float p0 = __expf(v0 - mnew);
      const float p1 = __expf(v1 - mnew);
      float sum = p0 + p1;
#pragma unroll
      for (int off = 1; off <= 8; off <<= 1)
        sum += __shfl_xor(sum, off, 32);
      l[r] = l[r] * corr + sum;
      m[r] = mnew;
#pragma unroll
      for (int db = 0; db < 4; ++db) acc[db][r] *= corr;
      // stage P tile (C-layout -> LDS row-major 16x32)
      ldsP[wave][(r + 8 * hi) * 32 + l15]      = (__bf16)p0;
      ldsP[wave][(r + 8 * hi) * 32 + 16 + l15] = (__bf16)p1;
    }

    asm volatile("s_wait_dscnt 0" ::: "memory");  // P stores visible to own wave

    // ---- reload P as A-fragment (row per lane), then O += P x V ----
    v16bf pa = ldbf16x16(&ldsP[wave][l15 * 32 + hi * 8]);
#pragma unroll
    for (int db = 0; db < 4; ++db) {
      v16bf vf = ldbf16x16(Vt2 + (db * 16 + l15) * 32 + hi * 8);
      acc[db] = wmma_bf16(pa, vf, acc[db]);
    }

    __syncthreads();                 // reads done before next stage overwrites
    buf ^= 1;
  }

  // ---- normalize and store token-major bf16 [B*S, 1024] ----
  const int b = bh >> 4, h = bh & 15;
#pragma unroll
  for (int r = 0; r < 8; ++r) {
    const float inv = 1.0f / l[r];
    const int srow = q0 + r + 8 * hi;
    const size_t rowoff = ((size_t)(b * SEQ_ + srow)) * EMB_ + h * DK_;
#pragma unroll
    for (int db = 0; db < 4; ++db)
      Abuf[rowoff + db * 16 + l15] = (__bf16)(acc[db][r] * inv);
  }
}

// ---------------------------------------------------------------------------
extern "C" void kernel_launch(void* const* d_in, const int* in_sizes, int n_in,
                              void* d_out, int out_size, void* d_ws, size_t ws_size,
                              hipStream_t stream) {
  const float* q  = (const float*)d_in[0];
  const float* k  = (const float*)d_in[1];
  const float* v  = (const float*)d_in[2];
  const float* Wq = (const float*)d_in[3];
  const float* bq = (const float*)d_in[4];
  const float* Wk = (const float*)d_in[5];
  const float* bk = (const float*)d_in[6];
  const float* Wv = (const float*)d_in[7];
  const float* bv = (const float*)d_in[8];
  const float* Wo = (const float*)d_in[9];
  const float* bo = (const float*)d_in[10];

  uint8_t* ws = (uint8_t*)d_ws;
  const size_t wBytes    = (size_t)EMB_ * EMB_ * sizeof(__bf16);                  // 2 MB
  const size_t xBytes    = (size_t)MROWS_ * EMB_ * sizeof(__bf16);                // 16 MB
  const size_t headBytes = (size_t)BATCH_ * NHEAD_ * SEQ_ * DK_ * sizeof(__bf16); // 16 MB

  __bf16* Wqb = (__bf16*)(ws);
  __bf16* Wkb = (__bf16*)(ws + 1 * wBytes);
  __bf16* Wvb = (__bf16*)(ws + 2 * wBytes);
  __bf16* Wob = (__bf16*)(ws + 3 * wBytes);
  uint8_t* base = ws + 4 * wBytes;

  // Weight conversion: 1024*1024/4 float4's per matrix
  const int w4 = (EMB_ * EMB_) / 4;
  cvt_f32_to_bf16<<<dim3(w4 / 256), dim3(256), 0, stream>>>(Wq, Wqb, w4);
  cvt_f32_to_bf16<<<dim3(w4 / 256), dim3(256), 0, stream>>>(Wk, Wkb, w4);
  cvt_f32_to_bf16<<<dim3(w4 / 256), dim3(256), 0, stream>>>(Wv, Wvb, w4);
  cvt_f32_to_bf16<<<dim3(w4 / 256), dim3(256), 0, stream>>>(Wo, Wob, w4);

  dim3 gemmGrid(EMB_ / 128, MROWS_ / 128);        // (8, 64)
  dim3 gemmBlk(256);
  dim3 faGrid(SEQ_ / 64, BATCH_ * NHEAD_);        // (32, 64)

  const size_t needBig = 4 * wBytes + 3 * xBytes + 3 * headBytes; // Abuf aliases Xq
  if (ws_size >= needBig) {
    // Big-ws path: pre-convert q/k/v to bf16 (amortized over 8 N-blocks each,
    // halves A-side read bytes, removes cvt VALU from the GEMM inner loop).
    __bf16* Xq = (__bf16*)(base);
    __bf16* Xk = (__bf16*)(base + xBytes);
    __bf16* Xv = (__bf16*)(base + 2 * xBytes);
    uint8_t* hbase = base + 3 * xBytes;
    __bf16* Qh   = (__bf16*)(hbase);
    __bf16* Kh   = (__bf16*)(hbase + headBytes);
    __bf16* Vt   = (__bf16*)(hbase + 2 * headBytes);
    __bf16* Abuf = Xq;   // Xq is dead after the Q projection

    const int x4 = (MROWS_ * EMB_) / 4;
    cvt_f32_to_bf16<<<dim3(x4 / 256), dim3(256), 0, stream>>>(q, Xq, x4);
    cvt_f32_to_bf16<<<dim3(x4 / 256), dim3(256), 0, stream>>>(k, Xk, x4);
    cvt_f32_to_bf16<<<dim3(x4 / 256), dim3(256), 0, stream>>>(v, Xv, x4);

    mha_gemm_wmma<<<gemmGrid, gemmBlk, 0, stream>>>(Xq, Wqb, bq, Qh, 0, 1);
    mha_gemm_wmma<<<gemmGrid, gemmBlk, 0, stream>>>(Xk, Wkb, bk, Kh, 0, 1);
    mha_gemm_wmma<<<gemmGrid, gemmBlk, 0, stream>>>(Xv, Wvb, bv, Vt, 2, 1);
    mha_flash_wmma<<<faGrid, dim3(128), 0, stream>>>(Qh, Kh, Vt, Abuf);
    mha_gemm_wmma<<<gemmGrid, gemmBlk, 0, stream>>>(Abuf, Wob, bo, d_out, 1, 1);
  } else {
    // Compact path: GEMM converts f32 A in-kernel.
    __bf16* Qh   = (__bf16*)(base);
    __bf16* Kh   = (__bf16*)(base + headBytes);
    __bf16* Vt   = (__bf16*)(base + 2 * headBytes);
    __bf16* Abuf = (__bf16*)(base + 3 * headBytes);

    mha_gemm_wmma<<<gemmGrid, gemmBlk, 0, stream>>>(q, Wqb, bq, Qh, 0, 0);
    mha_gemm_wmma<<<gemmGrid, gemmBlk, 0, stream>>>(k, Wkb, bk, Kh, 0, 0);
    mha_gemm_wmma<<<gemmGrid, gemmBlk, 0, stream>>>(v, Wvb, bv, Vt, 2, 0);
    mha_flash_wmma<<<faGrid, dim3(128), 0, stream>>>(Qh, Kh, Vt, Abuf);
    mha_gemm_wmma<<<gemmGrid, gemmBlk, 0, stream>>>(Abuf, Wob, bo, d_out, 1, 1);
  }
}
